// SelfAttHead_12567074308580
// MI455X (gfx1250) — matrix operand order repaired and checked by reading, not compiled, not run on hardware
//
#include <hip/hip_runtime.h>
#include <hip/hip_bf16.h>

typedef __bf16 bf16_t;
typedef __bf16 v16bf __attribute__((ext_vector_type(16)));
typedef __bf16 v8bf  __attribute__((ext_vector_type(8)));
typedef float  v8f   __attribute__((ext_vector_type(8)));

#define TM 128
#define TN 128
#define TK 64
#define LDSP 72  // 64 + 8 halves pad: 144B row stride, 16B-aligned, bank-conflict-free

// Async copy 64 contiguous bytes global -> LDS (4 x b128), no VGPR data staging.
// INST_OFFSET is added to BOTH the LDS and the global address (ISA 10.4), so one
// LDS-offset VGPR + one global-address VGPR pair covers all four chunks.
__device__ __forceinline__ void async_cp64(unsigned lds_off, const bf16_t* g) {
  asm volatile(
      "global_load_async_to_lds_b128 %0, %1, off\n\t"
      "global_load_async_to_lds_b128 %0, %1, off offset:16\n\t"
      "global_load_async_to_lds_b128 %0, %1, off offset:32\n\t"
      "global_load_async_to_lds_b128 %0, %1, off offset:48"
      :: "v"(lds_off), "v"(g) : "memory");
}
__device__ __forceinline__ void wait_async0() {
  asm volatile("s_wait_asynccnt 0x0" ::: "memory");
}

// Load a 16-half WMMA fragment from LDS as two 16B chunks separated by `gap` halves.
__device__ __forceinline__ v16bf frag_ld(const bf16_t* p, int gap) {
  v8bf lo = *(const v8bf*)p;
  v8bf hi = *(const v8bf*)(p + gap);
  v16bf r;
#pragma unroll
  for (int i = 0; i < 8; ++i) { r[i] = lo[i]; r[i + 8] = hi[i]; }
  return r;
}

enum { OUT_F32 = 0, OUT_BF16 = 1, OUT_BF16_T = 2 };

// C[M x N] = A[M x K] * Bt[N x K]^T, bf16 inputs, f32 accumulate via WMMA.
// Double-buffered LDS pipeline using CDNA5 async global->LDS DMA: tile k+1 is
// DMA'd into the alternate buffer (ASYNCcnt) while WMMAs consume tile k from
// LDS. One s_wait_asynccnt + barrier per K-tile; zero VGPRs spent on staging.
// OUT_BF16_T writes C transposed per batch: C[gm/tS][col][gm%tS] (used for v^T).
template <int OM>
__global__ __launch_bounds__(256) void gemm_wmma_bf16(
    const bf16_t* __restrict__ A, const bf16_t* __restrict__ Bt,
    void* __restrict__ Cv,
    int M, int N, int K, int lda, int ldb, int ldc,
    long abstr, long bbstr, long cbstr, int tS, int tH) {
  __shared__ bf16_t As[2][TM * LDSP];
  __shared__ bf16_t Bs[2][TN * LDSP];

  const int bz = blockIdx.z;
  A  += abstr * bz;
  Bt += bbstr * bz;
  const int m0   = blockIdx.y * TM;
  const int n0   = blockIdx.x * TN;
  const int tid  = threadIdx.x;
  const int lane = tid & 31;
  const int wid  = tid >> 5;
  const int wr   = wid >> 1;  // 0..3 -> 32-row slice
  const int wc   = wid & 1;   // 0..1 -> 64-col slice
  const int fm   = lane & 15; // row (A) / col (B) within 16x16 subtile
  const int kh   = lane >> 4; // lane-half selects K sub-chunk

  v8f acc[2][4];
#pragma unroll
  for (int i = 0; i < 2; ++i)
#pragma unroll
    for (int j = 0; j < 4; ++j)
#pragma unroll
      for (int r = 0; r < 8; ++r) acc[i][j][r] = 0.0f;

  const int lrow = tid >> 1;        // 0..127: tile row loaded by this thread
  const int lc   = (tid & 1) * 32;  // 0 or 32: which K-half of the row (64B slice)
  const bf16_t* gA = A  + (long)(m0 + lrow) * lda + lc;
  const bf16_t* gB = Bt + (long)(n0 + lrow) * ldb + lc;
  // Low 32 bits of a generic __shared__ pointer == LDS byte offset.
  const unsigned loA0 = (unsigned)(uintptr_t)&As[0][lrow * LDSP + lc];
  const unsigned loB0 = (unsigned)(uintptr_t)&Bs[0][lrow * LDSP + lc];
  const unsigned bufstep = (unsigned)(TM * LDSP * sizeof(bf16_t));

  auto async_fetch = [&](int k0, int b) {
    async_cp64(loA0 + (unsigned)b * bufstep, gA + k0);
    async_cp64(loB0 + (unsigned)b * bufstep, gB + k0);
  };
  auto compute = [&](const bf16_t* as, const bf16_t* bs) {
#pragma unroll
    for (int kk = 0; kk < TK; kk += 32) {
      v16bf af[2], bfv[4];
#pragma unroll
      for (int mt = 0; mt < 2; ++mt)  // A 16x32: halves 0-7 -> K+kh*8, 8-15 -> K+16+kh*8
        af[mt] = frag_ld(&as[(wr * 32 + mt * 16 + fm) * LDSP + kk + kh * 8], 16);
#pragma unroll
      for (int nt = 0; nt < 4; ++nt)  // B 32x16: halves 0-7 -> K+kh*8, 8-15 -> K+16+kh*8
        bfv[nt] = frag_ld(&bs[(wc * 64 + nt * 16 + fm) * LDSP + kk + kh * 8], 16);
#pragma unroll
      for (int mt = 0; mt < 2; ++mt)
#pragma unroll
        for (int nt = 0; nt < 4; ++nt)
          acc[mt][nt] = __builtin_amdgcn_wmma_f32_16x16x32_bf16(
              false, af[mt], false, bfv[nt], (short)0, acc[mt][nt], false, false);
    }
  };

  // Prologue: DMA tile 0 into buffer 0.
  async_fetch(0, 0);
  wait_async0();
  __syncthreads();

  int buf = 0;
  for (int k0 = TK; k0 < K; k0 += TK) {
    async_fetch(k0, buf ^ 1);        // DMA next tile into alternate buffer...
    compute(As[buf], Bs[buf]);       // ...while computing current tile from LDS
    wait_async0();                   // DMA done
    __syncthreads();                 // visible to all waves; one barrier per tile
    buf ^= 1;
  }
  compute(As[buf], Bs[buf]);         // epilogue tile

#pragma unroll
  for (int mt = 0; mt < 2; ++mt) {
#pragma unroll
    for (int nt = 0; nt < 4; ++nt) {
      const int col  = n0 + wc * 64 + nt * 16 + fm;
      const int rowb = m0 + wr * 32 + mt * 16 + kh * 8;  // C layout: VGPR r -> row rowb+r
      if (OM == OUT_F32) {
        float* C = (float*)Cv + cbstr * bz;
#pragma unroll
        for (int r = 0; r < 8; ++r) C[(long)(rowb + r) * ldc + col] = acc[mt][nt][r];
      } else if (OM == OUT_BF16) {
        bf16_t* C = (bf16_t*)Cv + cbstr * bz;
#pragma unroll
        for (int r = 0; r < 8; ++r) C[(long)(rowb + r) * ldc + col] = (bf16_t)acc[mt][nt][r];
      } else {  // transposed bf16 store: C[b][col][s], b = gm/tS, s = gm%tS
        bf16_t* C = (bf16_t*)Cv;
#pragma unroll
        for (int r = 0; r < 8; ++r) {
          int gm = rowb + r;
          C[(long)(gm / tS) * tH * tS + (long)col * tS + (gm % tS)] = (bf16_t)acc[mt][nt][r];
        }
      }
    }
  }
}

__global__ __launch_bounds__(256) void cvt_f32_bf16(const float* __restrict__ in,
                                                    bf16_t* __restrict__ out, long n) {
  long i = ((long)blockIdx.x * 256 + threadIdx.x) * 4;
  if (i + 3 < n) {
    float4 f = *(const float4*)(in + i);
    out[i + 0] = (bf16_t)f.x;
    out[i + 1] = (bf16_t)f.y;
    out[i + 2] = (bf16_t)f.z;
    out[i + 3] = (bf16_t)f.w;
  }
}

// Softmax over axis=1 (the i index) of scores[b, i, j]: one thread per column j.
// Consecutive threads read consecutive j -> every pass is fully coalesced.
__global__ __launch_bounds__(256) void softmax_col(const float* __restrict__ scores,
                                                   bf16_t* __restrict__ ak, int S) {
  const int j = blockIdx.x * 256 + threadIdx.x;
  const long base = (long)blockIdx.y * S * S + j;
  float m = -INFINITY;
  for (int i = 0; i < S; ++i) m = fmaxf(m, scores[base + (long)i * S]);
  float sum = 0.0f;
  for (int i = 0; i < S; ++i) sum += __expf(scores[base + (long)i * S] - m);
  const float inv = 1.0f / sum;
  for (int i = 0; i < S; ++i)
    ak[base + (long)i * S] = (bf16_t)(__expf(scores[base + (long)i * S] - m) * inv);
}

extern "C" void kernel_launch(void* const* d_in, const int* in_sizes, int n_in,
                              void* d_out, int out_size, void* d_ws, size_t ws_size,
                              hipStream_t stream) {
  (void)in_sizes; (void)n_in; (void)out_size; (void)ws_size;
  const int B = 32, S = 1024, E = 768, H = 256;
  const float* x   = (const float*)d_in[0];
  const float* Wqk = (const float*)d_in[1];
  const float* Wvk = (const float*)d_in[2];
  float* hk = (float*)d_out;

  char* p = (char*)d_ws;
  bf16_t* xb   = (bf16_t*)p; p += (size_t)B * S * E * 2;  // 48 MB
  bf16_t* wqkb = (bf16_t*)p; p += (size_t)E * E * 2;      // 1.1 MB
  bf16_t* wvkb = (bf16_t*)p; p += (size_t)H * E * 2;      // 0.4 MB
  bf16_t* qeb  = (bf16_t*)p; p += (size_t)B * S * E * 2;  // 48 MB
  bf16_t* vt   = (bf16_t*)p; p += (size_t)B * H * S * 2;  // 16 MB  (v transposed: [B,H,S])
  float*  sc   = (float*)p;  p += (size_t)B * S * S * 4;  // 128 MB
  bf16_t* ak   = (bf16_t*)p;                              // 64 MB

  // Convert inputs to bf16 (all element counts divisible by 1024).
  cvt_f32_bf16<<<(int)(((long)B * S * E) / 1024), 256, 0, stream>>>(x, xb, (long)B * S * E);
  cvt_f32_bf16<<<(E * E) / 1024, 256, 0, stream>>>(Wqk, wqkb, (long)E * E);
  cvt_f32_bf16<<<(H * E) / 1024, 256, 0, stream>>>(Wvk, wvkb, (long)H * E);

  // Stage 1: qe = x @ W_qk^T   [B*S, E] x [E, E]^T -> bf16
  gemm_wmma_bf16<OUT_BF16><<<dim3(E / TN, (B * S) / TM, 1), 256, 0, stream>>>(
      xb, wqkb, qeb, B * S, E, E, E, E, E, 0, 0, 0, 0, 0);

  // Stage 2: v^T = (x @ W_vk^T)^T  -> vt[b, h, s] bf16 (transposed store)
  gemm_wmma_bf16<OUT_BF16_T><<<dim3(H / TN, (B * S) / TM, 1), 256, 0, stream>>>(
      xb, wvkb, vt, B * S, H, E, E, E, 0, 0, 0, 0, S, H);

  // Stage 3: scores[b] = x[b] @ qe[b]^T -> f32
  gemm_wmma_bf16<OUT_F32><<<dim3(S / TN, S / TM, B), 256, 0, stream>>>(
      xb, qeb, sc, S, S, E, E, E, S, (long)S * E, (long)S * E, (long)S * S, 0, 0);

  // Stage 4: column softmax (axis=1) -> ak bf16
  softmax_col<<<dim3(S / 256, B), 256, 0, stream>>>(sc, ak, S);

  // Stage 5: hk[b] = ak[b] @ vt[b]^T  (= ak @ v) -> f32 output
  gemm_wmma_bf16<OUT_F32><<<dim3(H / TN, S / TM, B), 256, 0, stream>>>(
      ak, vt, hk, S, H, S, S, S, H, (long)S * S, (long)H * S, (long)S * H, 0, 0);
}
